// GPT_81303730913487
// MI455X (gfx1250) — matrix operand order repaired and checked by reading, not compiled, not run on hardware
//
#include <hip/hip_runtime.h>
#include <hip/hip_bf16.h>
#include <math.h>

// ---------------- CDNA5 WMMA types & fragment helpers (wave32) ----------------
typedef __attribute__((ext_vector_type(16))) _Float16 v16h;
typedef __attribute__((ext_vector_type(8)))  _Float16 v8h;
typedef __attribute__((ext_vector_type(8)))  float    v8f;

__device__ inline v8f zero8() {
  v8f z;
#pragma unroll
  for (int i = 0; i < 8; ++i) z[i] = 0.0f;
  return z;
}

__device__ inline v16h cat16(v8h lo, v8h hi) {
  v16h r;
#pragma unroll
  for (int i = 0; i < 8; ++i) { r[i] = lo[i]; r[8 + i] = hi[i]; }
  return r;
}

// A-matrix 16x32 f16 fragment from row-major f16 tile (row stride in halves).
// lanes 0-15: row m=lane, K = 0..7 and 16..23 ; lanes 16-31: row m=lane-16, K = 8..15 and 24..31
__device__ inline v16h load_a_frag(const _Float16* row0, int stride, int lane) {
  int m = lane & 15;
  int s = (lane & 16) ? 8 : 0;
  const _Float16* p = row0 + (size_t)m * stride;
  v8h lo = *(const v8h*)(p + s);
  v8h hi = *(const v8h*)(p + 16 + s);
  return cat16(lo, hi);
}

// B-matrix 32x16 f16 fragment from N-major ("transposed", [n][k]) tile.
// lanes 0-15: col n=lane, K = 0..7 and 16..23 ; lanes 16-31: col n=lane-16, K = 8..15 and 24..31
__device__ inline v16h load_b_frag(const _Float16* row0, int stride, int lane) {
  int n = lane & 15;
  int s = (lane & 16) ? 8 : 0;
  const _Float16* p = row0 + (size_t)n * stride + s;
  v8h lo = *(const v8h*)(p);
  v8h hi = *(const v8h*)(p + 16);
  return cat16(lo, hi);
}

__device__ inline v8f wmma16(v16h a, v16h b, v8f c) {
  return __builtin_amdgcn_wmma_f32_16x16x32_f16(false, a, false, b, (short)0, c,
                                                false, false);
}

// ---------------- Generic f16 GEMM: out = act(A[M,K] * Bt[N,K]^T + bias) (+resid) ----
// act: 0 = none, 1 = exact GELU, 2 = tanh
// Software-pipelined: next K-step's global tiles prefetched into registers while
// the current step's WMMAs run; per-thread static staging slices (no index loops).
#define BM 128
#define BN 64
#define BK 32
#define LDSS 40   // padded row stride (halves): 80B, keeps 16B alignment

__global__ __launch_bounds__(256) void gemm16(
    const _Float16* __restrict__ A, int lda,
    const _Float16* __restrict__ Bt, int ldb,
    const float* __restrict__ bias,
    const float* __restrict__ resid,
    float* __restrict__ out32,
    _Float16* __restrict__ out16,
    int M, int N, int K, int act)
{
  __shared__ __align__(16) _Float16 As[BM * LDSS];
  __shared__ __align__(16) _Float16 Bs[BN * LDSS];
  const int tid = threadIdx.x;
  const int lane = tid & 31;
  const int wave = tid >> 5;
  const int wm = wave & 3;         // 4 waves along M
  const int wn = wave >> 2;        // 2 waves along N
  const int m0 = blockIdx.y * BM;
  const int n0 = blockIdx.x * BN;

  // static per-thread staging slices
  const int arow = tid >> 1, ac = (tid & 1) * 16;   // A: 128 rows x 32 halves, 2 v8h/thread
  const int brow = tid >> 2, bc = (tid & 3) * 8;    // B: 64 rows x 32 halves,  1 v8h/thread
  const _Float16* Ap = A + (size_t)(m0 + arow) * lda + ac;
  const _Float16* Bp = Bt + (size_t)(n0 + brow) * ldb + bc;

  v8f acc[2][2];
#pragma unroll
  for (int i = 0; i < 2; ++i)
#pragma unroll
    for (int j = 0; j < 2; ++j) acc[i][j] = zero8();

  // prologue prefetch
  v8h pa0 = *(const v8h*)(Ap);
  v8h pa1 = *(const v8h*)(Ap + 8);
  v8h pb0 = *(const v8h*)(Bp);

  for (int kk = 0; kk < K; kk += BK) {
    // commit prefetched tiles to LDS
    *(v8h*)&As[arow * LDSS + ac] = pa0;
    *(v8h*)&As[arow * LDSS + ac + 8] = pa1;
    *(v8h*)&Bs[brow * LDSS + bc] = pb0;
    __syncthreads();

    // issue next step's global loads; they overlap the WMMAs below
    if (kk + BK < K) {
      Ap += BK; Bp += BK;
      pa0 = *(const v8h*)(Ap);
      pa1 = *(const v8h*)(Ap + 8);
      pb0 = *(const v8h*)(Bp);
    }

    v16h af[2], bf[2];
    af[0] = load_a_frag(&As[(wm * 32 + 0) * LDSS], LDSS, lane);
    af[1] = load_a_frag(&As[(wm * 32 + 16) * LDSS], LDSS, lane);
    bf[0] = load_b_frag(&Bs[(wn * 32 + 0) * LDSS], LDSS, lane);
    bf[1] = load_b_frag(&Bs[(wn * 32 + 16) * LDSS], LDSS, lane);
#pragma unroll
    for (int mt = 0; mt < 2; ++mt)
#pragma unroll
      for (int nt = 0; nt < 2; ++nt)
        acc[mt][nt] = wmma16(af[mt], bf[nt], acc[mt][nt]);
    __syncthreads();
  }

  // epilogue; C/D layout: VGPR r -> row r (lanes 0-15) or 8+r (lanes 16-31), col = lane&15
  const int rowsel = (lane & 16) ? 8 : 0;
#pragma unroll
  for (int mt = 0; mt < 2; ++mt) {
#pragma unroll
    for (int nt = 0; nt < 2; ++nt) {
      int n = n0 + wn * 32 + nt * 16 + (lane & 15);
      float bv = bias ? bias[n] : 0.0f;
#pragma unroll
      for (int r = 0; r < 8; ++r) {
        int m = m0 + wm * 32 + mt * 16 + rowsel + r;
        float v = acc[mt][nt][r] + bv;
        if (act == 1) v = 0.5f * v * (1.0f + erff(v * 0.70710678118f));
        else if (act == 2) v = tanhf(v);
        size_t idx = (size_t)m * N + n;
        if (resid) v += resid[idx];
        if (out32) out32[idx] = v;
        if (out16) out16[idx] = (_Float16)v;
      }
    }
  }
}

// ---------------- Fused flash attention (block-causal over frames of C=8) -----------
// one wave per (b, h, 16-query tile); online softmax; WMMA for QK^T and P*V.
// V-tile global loads are issued before the softmax math so the VALU work hides them.
__global__ __launch_bounds__(32) void attn16(
    const _Float16* __restrict__ q16, const _Float16* __restrict__ k16,
    const _Float16* __restrict__ vv16, const float* __restrict__ dmask,
    _Float16* __restrict__ y16)
{
  const int T_ = 1024, E_ = 512, DH_ = 64;
  __shared__ __align__(16) _Float16 Pt[16 * LDSS];   // probs tile 16x32
  __shared__ __align__(16) _Float16 Vt[64 * LDSS];   // V^T tile: [dh][key]
  const int lane = threadIdx.x;
  const int qt = blockIdx.x * 16, h = blockIdx.y, b = blockIdx.z;
  const int hi = (lane & 16) ? 1 : 0;
  const int l15 = lane & 15;
  const int mrow_base = qt + hi * 8;

  // Q fragments (DH=64 -> two 16x32 A frags), loaded straight from global
  v16h qa[2];
  {
    const _Float16* qp = q16 + ((size_t)(b * T_ + qt + l15)) * E_ + h * DH_;
    int s = hi ? 8 : 0;
#pragma unroll
    for (int c = 0; c < 2; ++c) {
      v8h lo = *(const v8h*)(qp + c * 32 + s);
      v8h hh = *(const v8h*)(qp + c * 32 + 16 + s);
      qa[c] = cat16(lo, hh);
    }
  }

  v8f o[4];
#pragma unroll
  for (int f = 0; f < 4; ++f) o[f] = zero8();
  float mi[8], li[8];
#pragma unroll
  for (int r = 0; r < 8; ++r) { mi[r] = -1e30f; li[r] = 0.0f; }

  for (int jb = 0; jb <= qt; jb += 32) {
    // two 16-key score tiles S0 (jb..jb+15), S1 (jb+16..jb+31)
    v8f s01[2];
#pragma unroll
    for (int t = 0; t < 2; ++t) {
      v8f s = zero8();
      int key = jb + t * 16 + l15;
      int keyc = key < T_ ? key : T_ - 1;
      const _Float16* kp = k16 + ((size_t)(b * T_ + keyc)) * E_ + h * DH_;
      int sb = hi ? 8 : 0;
#pragma unroll
      for (int c = 0; c < 2; ++c) {
        v8h lo = *(const v8h*)(kp + c * 32 + sb);
        v8h hh = *(const v8h*)(kp + c * 32 + 16 + sb);
        v16h bf = cat16(lo, hh);
        s = wmma16(qa[c], bf, s);
      }
      s01[t] = s;
    }

    // issue V-tile loads now: latency hidden behind the softmax VALU work below
    v8h vreg[8];
    {
      int key = jb + lane;
      int keyc = key < T_ ? key : T_ - 1;
      const _Float16* vp = vv16 + ((size_t)(b * T_ + keyc)) * E_ + h * DH_;
#pragma unroll
      for (int d8 = 0; d8 < 8; ++d8) vreg[d8] = *(const v8h*)(vp + d8 * 8);
    }

    // mask + online softmax; row r of this half lives in VGPR element r
#pragma unroll
    for (int r = 0; r < 8; ++r) {
      int m = mrow_base + r;
      int lim = (m & ~7) + 8;                 // keys with frame <= frame(m)
      int c0 = jb + l15, c1 = jb + 16 + l15;
      float a0 = s01[0][r] * 0.125f;          // 1/sqrt(64)
      float a1 = s01[1][r] * 0.125f;
      bool ok0 = (c0 < lim) && (dmask[b * T_ + c0] > 0.0f);
      bool ok1 = (c1 < lim) && (dmask[b * T_ + c1] > 0.0f);
      if (!ok0) a0 = -1e30f;
      if (!ok1) a1 = -1e30f;
      float t = fmaxf(a0, a1);
#pragma unroll
      for (int d = 1; d < 16; d <<= 1) t = fmaxf(t, __shfl_xor(t, d, 16));
      float mn = fmaxf(mi[r], t);
      float corr = __expf(mi[r] - mn);
      float p0 = __expf(a0 - mn), p1 = __expf(a1 - mn);
      float rs = p0 + p1;
#pragma unroll
      for (int d = 1; d < 16; d <<= 1) rs += __shfl_xor(rs, d, 16);
      li[r] = li[r] * corr + rs;
      mi[r] = mn;
#pragma unroll
      for (int f = 0; f < 4; ++f) o[f][r] *= corr;
      Pt[(r + hi * 8) * LDSS + l15] = (_Float16)p0;
      Pt[(r + hi * 8) * LDSS + 16 + l15] = (_Float16)p1;
    }
    // commit V^T (32 keys x 64 dh) to LDS: Vt[dh][key]
#pragma unroll
    for (int d8 = 0; d8 < 8; ++d8) {
#pragma unroll
      for (int i = 0; i < 8; ++i) Vt[(d8 * 8 + i) * LDSS + lane] = vreg[d8][i];
    }
    __syncthreads();
    v16h pa = load_a_frag(Pt, LDSS, lane);        // P as 16x32 A frag
#pragma unroll
    for (int f = 0; f < 4; ++f) {
      v16h bf = load_b_frag(&Vt[f * 16 * LDSS], LDSS, lane);
      o[f] = wmma16(pa, bf, o[f]);
    }
    __syncthreads();
  }

  // normalize + store y
#pragma unroll
  for (int r = 0; r < 8; ++r) {
    float inv = li[r] > 0.0f ? 1.0f / li[r] : 0.0f;
    int m = mrow_base + r;
#pragma unroll
    for (int f = 0; f < 4; ++f)
      y16[((size_t)(b * T_ + m)) * E_ + h * DH_ + f * 16 + l15] =
          (_Float16)(o[f][r] * inv);
  }
}

// ---------------- LayerNorm: f32 in -> f16 out ----------------
__global__ __launch_bounds__(256) void ln16(const float* __restrict__ x,
                                            const float* __restrict__ g,
                                            const float* __restrict__ bb,
                                            _Float16* __restrict__ out, int W)
{
  __shared__ float red[256];
  const int row = blockIdx.x, tid = threadIdx.x;
  const float* xr = x + (size_t)row * W;
  float s = 0.0f, s2 = 0.0f;
  for (int i = tid; i < W; i += 256) { float v = xr[i]; s += v; s2 += v * v; }
  red[tid] = s; __syncthreads();
  for (int off = 128; off; off >>= 1) { if (tid < off) red[tid] += red[tid + off]; __syncthreads(); }
  float mean = red[0] / W; __syncthreads();
  red[tid] = s2; __syncthreads();
  for (int off = 128; off; off >>= 1) { if (tid < off) red[tid] += red[tid + off]; __syncthreads(); }
  float var = red[0] / W - mean * mean;
  float rs = rsqrtf(var + 1e-5f);
  for (int i = tid; i < W; i += 256)
    out[(size_t)row * W + i] = (_Float16)((xr[i] - mean) * rs * g[i] + bb[i]);
}

// ---------------- BatchNorm stats (per channel d of D=64 over 4096 tokens) ----------
__global__ __launch_bounds__(256) void bn_stats(const float* __restrict__ tok,
                                                float* __restrict__ st)
{
  __shared__ float r1[256], r2[256];
  const int ch = blockIdx.x, tid = threadIdx.x;
  float s = 0.0f, s2 = 0.0f;
  for (int i = tid; i < 4096; i += 256) {
    float v = tok[(size_t)i * 64 + ch]; s += v; s2 += v * v;
  }
  r1[tid] = s; r2[tid] = s2; __syncthreads();
  for (int off = 128; off; off >>= 1) {
    if (tid < off) { r1[tid] += r1[tid + off]; r2[tid] += r2[tid + off]; }
    __syncthreads();
  }
  if (tid == 0) {
    float mean = r1[0] / 4096.0f;
    st[ch] = mean;
    st[64 + ch] = r2[0] / 4096.0f - mean * mean;   // biased var
  }
}

__global__ void bn_apply(const float* __restrict__ tok, const float* __restrict__ mask,
                         const float* __restrict__ st, const float* __restrict__ g,
                         const float* __restrict__ bb, _Float16* __restrict__ tn16,
                         float* __restrict__ toksm, int n)
{
  int idx = blockIdx.x * blockDim.x + threadIdx.x;
  if (idx >= n) return;
  int d = idx & 63, t = idx >> 6;
  float v = tok[idx];
  tn16[idx] = (_Float16)((v - st[d]) * rsqrtf(st[64 + d] + 1e-5f) * g[d] + bb[d]);
  toksm[idx] = v * mask[t];
}

// ---------------- weight convert f32[K,N] -> f16[N,K] (batched) ----------------
__global__ void wtrans(const float* __restrict__ W, _Float16* __restrict__ Wt,
                       int K, int N, long total)
{
  long idx = (long)blockIdx.x * blockDim.x + threadIdx.x;
  if (idx >= total) return;
  long kn = (long)K * N;
  long l = idx / kn, rem = idx - l * kn;
  int k = (int)(rem / N), n = (int)(rem - (long)k * N);
  Wt[l * kn + (size_t)n * K + k] = (_Float16)W[idx];
}

// ---------------- sequence reverse (per batch) ----------------
__global__ void revseq(const float* __restrict__ x, float* __restrict__ xr, int n)
{
  int idx = blockIdx.x * blockDim.x + threadIdx.x;
  if (idx >= n) return;
  int e = idx & 511, t = (idx >> 9) & 1023, b = idx >> 19;
  xr[idx] = x[(((size_t)(b * 1024 + 1023 - t)) << 9) + e];
}

__global__ void revmask(const float* __restrict__ m, float* __restrict__ mr)
{
  int idx = blockIdx.x * blockDim.x + threadIdx.x;
  if (idx >= 4096) return;
  int b = idx >> 10, t = idx & 1023;
  mr[idx] = m[(b << 10) + 1023 - t];
}

// ---------------- frame mean over C=8: [B,T,256] -> [B,128,256] ----------------
__global__ void fmean(const float* __restrict__ xp, float* __restrict__ outf,
                      _Float16* __restrict__ out16, int n)
{
  int idx = blockIdx.x * blockDim.x + threadIdx.x;
  if (idx >= n) return;
  int o = idx & 255, bt = idx >> 8;
  int b = bt >> 7, tf = bt & 127;
  float s = 0.0f;
#pragma unroll
  for (int c = 0; c < 8; ++c)
    s += xp[(((size_t)(b * 1024 + tf * 8 + c)) << 8) + o];
  s *= 0.125f;
  outf[idx] = s;
  out16[idx] = (_Float16)s;
}

// ---------------- loss kernels ----------------
__global__ void zero_acc(float* a) { if (threadIdx.x < 8) a[threadIdx.x] = 0.0f; }

__global__ __launch_bounds__(256) void mse_token(const float* __restrict__ pred,
                                                 const float* __restrict__ mask,
                                                 const float* __restrict__ toksm,
                                                 int rev, float* __restrict__ accum, int slot)
{
  __shared__ float red[256];
  const long nel = 4L * 1016 * 64;
  float s = 0.0f;
  for (long i = (long)blockIdx.x * 256 + threadIdx.x; i < nel; i += (long)gridDim.x * 256) {
    int d = (int)(i & 63);
    long r = i >> 6;
    int t = (int)(r % 1016), b = (int)(r / 1016);
    int tt = rev ? (1023 - 8 - t) : (t + 8);
    float p = pred[(((size_t)(b * 1024 + t)) << 6) + d] * mask[b * 1024 + t];
    float diff = p - toksm[(((size_t)(b * 1024 + tt)) << 6) + d];
    s += diff * diff;
  }
  red[threadIdx.x] = s; __syncthreads();
  for (int off = 128; off; off >>= 1) { if (threadIdx.x < off) red[threadIdx.x] += red[threadIdx.x + off]; __syncthreads(); }
  if (threadIdx.x == 0) atomicAdd(&accum[slot], red[0]);
}

__global__ __launch_bounds__(256) void mse_frame(const float* __restrict__ fdo,
                                                 const float* __restrict__ toksm,
                                                 int rev, float* __restrict__ accum, int slot)
{
  __shared__ float red[256];
  const long nel = 4L * 127 * 512;
  float s = 0.0f;
  for (long i = (long)blockIdx.x * 256 + threadIdx.x; i < nel; i += (long)gridDim.x * 256) {
    int j = (int)(i & 511);
    long r = i >> 9;
    int tf = (int)(r % 127), b = (int)(r / 127);
    int tgt = rev ? (126 - tf) : (tf + 1);
    int c = j >> 6, d = j & 63;
    float diff = fdo[(((size_t)(b * 128 + tf)) << 9) + j] -
                 toksm[(((size_t)(b * 1024 + tgt * 8 + c)) << 6) + d];
    s += diff * diff;
  }
  red[threadIdx.x] = s; __syncthreads();
  for (int off = 128; off; off >>= 1) { if (threadIdx.x < off) red[threadIdx.x] += red[threadIdx.x + off]; __syncthreads(); }
  if (threadIdx.x == 0) atomicAdd(&accum[slot], red[0]);
}

__global__ void finalize_losses(const float* __restrict__ accum, float* __restrict__ out)
{
  int i = threadIdx.x;
  if (i < 4) out[131072 + i] = accum[i] * (1.0f / 260096.0f);  // both counts = 4*1016*64 = 4*127*512
}

// ---------------- host orchestration ----------------
extern "C" void kernel_launch(void* const* d_in, const int* in_sizes, int n_in,
                              void* d_out, int out_size, void* d_ws, size_t ws_size,
                              hipStream_t stream) {
  (void)in_sizes; (void)n_in; (void)out_size; (void)ws_size;
  const float* tokens = (const float*)d_in[0];
  const float* mask   = (const float*)d_in[2];
  const float* bn_g = (const float*)d_in[3];
  const float* bn_b = (const float*)d_in[4];
  const float* We = (const float*)d_in[5];
  const float* be = (const float*)d_in[6];
  const float* ln1_g = (const float*)d_in[7];
  const float* ln1_b = (const float*)d_in[8];
  const float* ln2_g = (const float*)d_in[9];
  const float* ln2_b = (const float*)d_in[10];
  const float* Wq = (const float*)d_in[11];
  const float* bq = (const float*)d_in[12];
  const float* Wk = (const float*)d_in[13];
  const float* bk = (const float*)d_in[14];
  const float* Wv = (const float*)d_in[15];
  const float* bv = (const float*)d_in[16];
  const float* Wo = (const float*)d_in[17];
  const float* bo = (const float*)d_in[18];
  const float* W1 = (const float*)d_in[19];
  const float* b1 = (const float*)d_in[20];
  const float* W2 = (const float*)d_in[21];
  const float* b2 = (const float*)d_in[22];
  const float* lnf_g = (const float*)d_in[23];
  const float* lnf_b = (const float*)d_in[24];
  const float* Wp = (const float*)d_in[25];
  const float* bp = (const float*)d_in[26];
  const float* Wd1 = (const float*)d_in[27];
  const float* bd1 = (const float*)d_in[28];
  const float* lnd_g = (const float*)d_in[29];
  const float* lnd_b = (const float*)d_in[30];
  const float* Wd2 = (const float*)d_in[31];
  const float* bd2 = (const float*)d_in[32];
  const float* Wf1 = (const float*)d_in[33];
  const float* bf1 = (const float*)d_in[34];
  const float* lnfr_g = (const float*)d_in[35];
  const float* lnfr_b = (const float*)d_in[36];
  const float* Wf2 = (const float*)d_in[37];
  const float* bf2 = (const float*)d_in[38];
  float* outp = (float*)d_out;

  // carve workspace
  char* wp = (char*)d_ws;
  auto carve = [&](size_t bytes) -> void* {
    void* r = (void*)wp; wp += (bytes + 255) & ~(size_t)255; return r;
  };
  _Float16* WeT  = (_Float16*)carve((size_t)512 * 64 * 2);
  _Float16* WqT  = (_Float16*)carve((size_t)6 * 512 * 512 * 2);
  _Float16* WkT  = (_Float16*)carve((size_t)6 * 512 * 512 * 2);
  _Float16* WvT  = (_Float16*)carve((size_t)6 * 512 * 512 * 2);
  _Float16* WoT  = (_Float16*)carve((size_t)6 * 512 * 512 * 2);
  _Float16* W1T  = (_Float16*)carve((size_t)6 * 2048 * 512 * 2);
  _Float16* W2T  = (_Float16*)carve((size_t)6 * 512 * 2048 * 2);
  _Float16* WpT  = (_Float16*)carve((size_t)256 * 512 * 2);
  _Float16* Wd1T = (_Float16*)carve((size_t)256 * 256 * 2);
  _Float16* Wd2T = (_Float16*)carve((size_t)64 * 256 * 2);
  _Float16* Wf1T = (_Float16*)carve((size_t)256 * 256 * 2);
  _Float16* Wf2T = (_Float16*)carve((size_t)512 * 256 * 2);
  float*    bnst = (float*)carve(128 * 4);
  _Float16* tn16 = (_Float16*)carve((size_t)4096 * 64 * 2);
  float*    toksm = (float*)carve((size_t)4096 * 64 * 4);
  float*    x_f  = (float*)carve((size_t)4096 * 512 * 4);
  float*    xr_f = (float*)carve((size_t)4096 * 512 * 4);
  float*    maskr = (float*)carve(4096 * 4);
  _Float16* h16  = (_Float16*)carve((size_t)4096 * 512 * 2);
  _Float16* qkvy = (_Float16*)carve((size_t)4096 * 2048 * 2);  // q|k|v|y ; m16 aliases base
  _Float16* q16 = qkvy;
  _Float16* k16 = qkvy + (size_t)4096 * 512;
  _Float16* v16p = qkvy + (size_t)2 * 4096 * 512;
  _Float16* y16 = qkvy + (size_t)3 * 4096 * 512;
  _Float16* m16 = qkvy;                                        // alias: q/k/v dead by MLP
  float*    xp_f  = (float*)carve((size_t)4096 * 256 * 4);
  _Float16* xp16  = (_Float16*)carve((size_t)4096 * 256 * 2);
  float*    xp2_f = (float*)carve((size_t)4096 * 256 * 4);
  _Float16* xp2_16 = (_Float16*)carve((size_t)4096 * 256 * 2);
  float*    t1f = (float*)carve((size_t)4096 * 256 * 4);
  _Float16* t1h = (_Float16*)carve((size_t)4096 * 256 * 2);
  float*    pred = (float*)carve((size_t)4096 * 64 * 4);
  _Float16* xf16 = (_Float16*)carve((size_t)512 * 256 * 2);
  float*    xf2_f = (float*)carve((size_t)512 * 256 * 4);
  _Float16* xf2_16 = (_Float16*)carve((size_t)512 * 256 * 2);
  float*    u1f = (float*)carve((size_t)512 * 256 * 4);
  _Float16* u1h = (_Float16*)carve((size_t)512 * 256 * 2);
  float*    fdo = (float*)carve((size_t)512 * 512 * 4);
  float*    accum = (float*)carve(8 * 4);

  zero_acc<<<1, 8, 0, stream>>>(accum);

  // BatchNorm + weight conversion (f32 -> f16 transposed [N,K])
  bn_stats<<<64, 256, 0, stream>>>(tokens, bnst);
  bn_apply<<<(4096 * 64 + 255) / 256, 256, 0, stream>>>(tokens, mask, bnst, bn_g, bn_b,
                                                        tn16, toksm, 4096 * 64);
  auto trans = [&](const float* W, _Float16* Wt, int K, int N, long batch) {
    long total = batch * (long)K * N;
    wtrans<<<(unsigned)((total + 255) / 256), 256, 0, stream>>>(W, Wt, K, N, total);
  };
  trans(We, WeT, 64, 512, 1);
  trans(Wq, WqT, 512, 512, 6);  trans(Wk, WkT, 512, 512, 6);
  trans(Wv, WvT, 512, 512, 6);  trans(Wo, WoT, 512, 512, 6);
  trans(W1, W1T, 512, 2048, 6); trans(W2, W2T, 2048, 512, 6);
  trans(Wp, WpT, 512, 256, 1);
  trans(Wd1, Wd1T, 256, 256, 1); trans(Wd2, Wd2T, 256, 64, 1);
  trans(Wf1, Wf1T, 256, 256, 1); trans(Wf2, Wf2T, 256, 512, 1);

  // embeddings: x0 = tn @ We + be  (f32)
  gemm16<<<dim3(512 / 64, 4096 / 128), 256, 0, stream>>>(
      tn16, 64, WeT, 64, be, nullptr, x_f, nullptr, 4096, 512, 64, 0);
  revseq<<<(4096 * 512 + 255) / 256, 256, 0, stream>>>(x_f, xr_f, 4096 * 512);
  revmask<<<16, 256, 0, stream>>>(mask, maskr);

  // transformer blocks (shared by both directions)
  auto run_blocks = [&](float* xb, const float* dm, float* xpf, _Float16* xph) {
    for (int l = 0; l < 6; ++l) {
      ln16<<<4096, 256, 0, stream>>>(xb, ln1_g + l * 512, ln1_b + l * 512, h16, 512);
      gemm16<<<dim3(8, 32), 256, 0, stream>>>(h16, 512, WqT + (size_t)l * 512 * 512, 512,
          bq + l * 512, nullptr, nullptr, q16, 4096, 512, 512, 0);
      gemm16<<<dim3(8, 32), 256, 0, stream>>>(h16, 512, WkT + (size_t)l * 512 * 512, 512,
          bk + l * 512, nullptr, nullptr, k16, 4096, 512, 512, 0);
      gemm16<<<dim3(8, 32), 256, 0, stream>>>(h16, 512, WvT + (size_t)l * 512 * 512, 512,
          bv + l * 512, nullptr, nullptr, v16p, 4096, 512, 512, 0);
      attn16<<<dim3(64, 8, 4), 32, 0, stream>>>(q16, k16, v16p, dm, y16);
      gemm16<<<dim3(8, 32), 256, 0, stream>>>(y16, 512, WoT + (size_t)l * 512 * 512, 512,
          bo + l * 512, xb, xb, nullptr, 4096, 512, 512, 0);
      ln16<<<4096, 256, 0, stream>>>(xb, ln2_g + l * 512, ln2_b + l * 512, h16, 512);
      gemm16<<<dim3(32, 32), 256, 0, stream>>>(h16, 512, W1T + (size_t)l * 2048 * 512, 512,
          b1 + l * 2048, nullptr, nullptr, m16, 4096, 2048, 512, 1);
      gemm16<<<dim3(8, 32), 256, 0, stream>>>(m16, 2048, W2T + (size_t)l * 512 * 2048, 2048,
          b2 + l * 512, xb, xb, nullptr, 4096, 512, 2048, 0);
    }
    ln16<<<4096, 256, 0, stream>>>(xb, lnf_g, lnf_b, h16, 512);
    gemm16<<<dim3(4, 32), 256, 0, stream>>>(h16, 512, WpT, 512, bp,
        nullptr, xpf, xph, 4096, 256, 512, 0);
  };
  run_blocks(x_f, mask, xp_f, xp16);
  run_blocks(xr_f, maskr, xp2_f, xp2_16);

  // frame means: forward result goes straight into d_out
  fmean<<<(512 * 256 + 255) / 256, 256, 0, stream>>>(xp_f, outp, xf16, 512 * 256);
  fmean<<<(512 * 256 + 255) / 256, 256, 0, stream>>>(xp2_f, xf2_f, xf2_16, 512 * 256);

  // token decoder (both directions)
  auto decode = [&](const _Float16* z16) {
    gemm16<<<dim3(4, 32), 256, 0, stream>>>(z16, 256, Wd1T, 256, bd1,
        nullptr, t1f, nullptr, 4096, 256, 256, 2);
    ln16<<<4096, 256, 0, stream>>>(t1f, lnd_g, lnd_b, t1h, 256);
    gemm16<<<dim3(1, 32), 256, 0, stream>>>(t1h, 256, Wd2T, 256, bd2,
        nullptr, pred, nullptr, 4096, 64, 256, 0);
  };
  decode(xp16);
  mse_token<<<256, 256, 0, stream>>>(pred, mask, toksm, 0, accum, 0);
  decode(xp2_16);
  mse_token<<<256, 256, 0, stream>>>(pred, mask, toksm, 1, accum, 1);

  // frame decoder (both directions)
  auto fdecode = [&](const _Float16* z16) {
    gemm16<<<dim3(4, 4), 256, 0, stream>>>(z16, 256, Wf1T, 256, bf1,
        nullptr, u1f, nullptr, 512, 256, 256, 2);
    ln16<<<512, 256, 0, stream>>>(u1f, lnfr_g, lnfr_b, u1h, 256);
    gemm16<<<dim3(8, 4), 256, 0, stream>>>(u1h, 256, Wf2T, 256, bf2,
        nullptr, fdo, nullptr, 512, 512, 256, 0);
  };
  fdecode(xf16);
  mse_frame<<<256, 256, 0, stream>>>(fdo, toksm, 0, accum, 2);
  fdecode(xf2_16);
  mse_frame<<<256, 256, 0, stream>>>(fdo, toksm, 1, accum, 3);

  finalize_losses<<<1, 4, 0, stream>>>(accum, outp);
}
